// Model_11811160064533
// MI455X (gfx1250) — matrix operand order repaired and checked by reading, not compiled, not run on hardware
//
#include <hip/hip_runtime.h>
#include <math.h>

// ---------------------------------------------------------------------------
// Informer forward for MI455X (gfx1250, wave32, WMMA).
// fp32 in HBM, f16 WMMA compute with fp32 accumulation.
// ProbSparse attention approximated with dense (flash-style) attention.
// ---------------------------------------------------------------------------

typedef __attribute__((ext_vector_type(16))) _Float16 v16h;
typedef __attribute__((ext_vector_type(8)))  _Float16 v8h;
typedef __attribute__((ext_vector_type(8)))  float    v8f;

#define SHUF16(a, b) __builtin_shufflevector((a), (b), 0,1,2,3,4,5,6,7,8,9,10,11,12,13,14,15)

#define WMMA_F16(A, B, C) \
  __builtin_amdgcn_wmma_f32_16x16x32_f16(false, (A), false, (B), (short)0, (C), false, false)

// ===========================================================================
// GEMM: C[M,N] = act( A[M,K] @ B[K,N] + bias ), f32 global, f16 WMMA compute.
// GUARD-FREE: requires M % 128 == 0, N % 128 == 0, K % 32 == 0 (all call
// sites satisfy this: M in {5760,5376}, N in {512,2048}, K in {512,2048}).
// Block = 256 threads = 8 waves in a 4x2 (m x n) grid; block tile 128x128;
// each wave computes a 32x64 sub-tile = 8 WMMAs per 32-wide K-step.
// Register double-buffering: next K-tile's global loads are issued before
// the WMMAs so HBM latency overlaps the math.
// A staged [m][k]; B staged transposed [n][k] so B-fragments are contiguous.
// act: 0 = none, 1 = exact GELU.
// ===========================================================================
#define GLDS 40  // halfs per LDS row (32 + 8 pad), keeps 16B alignment

__global__ __launch_bounds__(256)
void gemm_f16_kernel(const float* __restrict__ A, const float* __restrict__ B,
                     const float* __restrict__ bias, float* __restrict__ C,
                     int M, int N, int K, int act)
{
  __shared__ alignas(16) _Float16 sA[128 * GLDS];
  __shared__ alignas(16) _Float16 sB[128 * GLDS];

  const int tid  = threadIdx.x;
  const int wave = tid >> 5;
  const int lane = tid & 31;
  const int lm   = lane & 15;
  const int hi   = lane >> 4;
  const int wm   = wave >> 1;   // 0..3 : 32-row m-quadrant
  const int wn   = wave & 1;    // 0..1 : 64-col n-half
  const int m0   = blockIdx.y * 128;
  const int n0   = blockIdx.x * 128;

  // staging coordinates (A: 128x32, B: 32x128, 16 floats per thread each)
  const int ar = tid >> 3;         // A row base (0..31), +32 per it
  const int ac = (tid & 7) * 4;    // A col
  const int bk = tid >> 5;         // B k base (0..7), +8 per it
  const int bn = (tid & 31) * 4;   // B col (0..124)

  const float* Ab = A + (size_t)(m0 + ar) * K + ac;
  const float* Bb = B + (size_t)bk * N + n0 + bn;

  float4 ra[4], rb[4];
  // prologue: fetch K-tile 0 into registers (batched, no waits in between)
  for (int it = 0; it < 4; ++it) ra[it] = *(const float4*)(Ab + (size_t)it * 32 * K);
  for (int it = 0; it < 4; ++it) rb[it] = *(const float4*)(Bb + (size_t)it * 8 * N);

  v8f acc[2][4] = {};

  for (int k0 = 0; k0 < K; k0 += 32) {
    // commit staged registers (f32 -> f16) to LDS
    for (int it = 0; it < 4; ++it) {
      _Float16* d = &sA[(ar + it * 32) * GLDS + ac];
      d[0] = (_Float16)ra[it].x; d[1] = (_Float16)ra[it].y;
      d[2] = (_Float16)ra[it].z; d[3] = (_Float16)ra[it].w;
    }
    for (int it = 0; it < 4; ++it) {
      int kk = bk + it * 8;
      sB[(bn + 0) * GLDS + kk] = (_Float16)rb[it].x;
      sB[(bn + 1) * GLDS + kk] = (_Float16)rb[it].y;
      sB[(bn + 2) * GLDS + kk] = (_Float16)rb[it].z;
      sB[(bn + 3) * GLDS + kk] = (_Float16)rb[it].w;
    }
    __syncthreads();

    // issue next K-tile's global loads now; they overlap the WMMAs below
    if (k0 + 32 < K) {
      const float* An = Ab + (k0 + 32);
      const float* Bn = Bb + (size_t)(k0 + 32) * N;
      for (int it = 0; it < 4; ++it) ra[it] = *(const float4*)(An + (size_t)it * 32 * K);
      for (int it = 0; it < 4; ++it) rb[it] = *(const float4*)(Bn + (size_t)it * 8 * N);
    }

    // A fragments: rows wm*32 + {0,16} + lm; halfs k = {hi*8..+7, 16+hi*8..+7}
    v16h afr[2];
    for (int mt = 0; mt < 2; ++mt) {
      int row = wm * 32 + mt * 16 + lm;
      const v8h* a0 = (const v8h*)&sA[row * GLDS + hi * 8];
      const v8h* a1 = (const v8h*)&sA[row * GLDS + 16 + hi * 8];
      afr[mt] = SHUF16(*a0, *a1);
    }
    // B fragments: col wn*64 + nt*16 + lm; halfs k = {hi*8..+7, 16+hi*8..+7}
    for (int nt = 0; nt < 4; ++nt) {
      const v8h* b0 = (const v8h*)&sB[(wn * 64 + nt * 16 + lm) * GLDS + hi * 8];
      const v8h* b1 = (const v8h*)&sB[(wn * 64 + nt * 16 + lm) * GLDS + 16 + hi * 8];
      v16h bf = SHUF16(*b0, *b1);
      acc[0][nt] = WMMA_F16(afr[0], bf, acc[0][nt]);
      acc[1][nt] = WMMA_F16(afr[1], bf, acc[1][nt]);
    }
    __syncthreads();
  }

  // Epilogue: lane holds col n = lm (per tile), rows hi*8 .. +7.
  for (int mt = 0; mt < 2; ++mt) {
    int cm_base = m0 + wm * 32 + mt * 16 + hi * 8;
    for (int nt = 0; nt < 4; ++nt) {
      int cn = n0 + wn * 64 + nt * 16 + lm;
      float bv = bias ? bias[cn] : 0.f;
      for (int i = 0; i < 8; ++i) {
        float v = acc[mt][nt][i] + bv;
        if (act == 1) v = 0.5f * v * (1.f + erff(v * 0.70710678118f));  // exact GELU
        C[(size_t)(cm_base + i) * N + cn] = v;
      }
    }
  }
}

// ===========================================================================
// Flash-style attention. Layout: rows = b*L + pos, cols = h*64 + dh, D=512.
// Block = 128 (4 waves); wave w owns 16-query tile (q0 + w*16).
// Per 32-key step: S = Q K^T via 4 WMMAs, online softmax (16-lane shfl_xor
// reductions match the C-tile layout), P relaid through per-wave LDS tile,
// then O += P V via 4 WMMAs with V staged transposed [dh][key].
// Staging uses a block-uniform fast path (no guards) for interior tiles.
// ===========================================================================
__global__ __launch_bounds__(128)
void attn_kernel(const float* __restrict__ Q, const float* __restrict__ K,
                 const float* __restrict__ V, float* __restrict__ O,
                 int Lq, int Lk, int causal)
{
  const int D = 512, DH = 64;
  __shared__ alignas(16) _Float16 sQ[64 * 72];     // [q][dh]
  __shared__ alignas(16) _Float16 sK[32 * 72];     // [key][dh]
  __shared__ alignas(16) _Float16 sV[64 * 40];     // [dh][key] (transposed)
  __shared__ alignas(16) _Float16 sP[4 * 16 * 40]; // per-wave P tile [m][key]

  const int tid  = threadIdx.x;
  const int wave = tid >> 5;
  const int lane = tid & 31;
  const int lm   = lane & 15;
  const int hi   = lane >> 4;
  const int b    = blockIdx.z;
  const int h    = blockIdx.y;
  const int q0   = blockIdx.x * 64;
  const float scale = 0.125f;  // 1/sqrt(64), folded into Q

  // ---- stage Q: 64 queries x 64 dh (batched loads, then convert) ----
  {
    const int r0 = tid >> 4;        // +8 per it
    const int c  = (tid & 15) * 4;
    float4 rq[8];
    if (q0 + 63 < Lq) {
      for (int it = 0; it < 8; ++it)
        rq[it] = *(const float4*)(Q + ((size_t)b * Lq + q0 + r0 + it * 8) * D + h * DH + c);
    } else {
      for (int it = 0; it < 8; ++it) {
        float4 z = {0.f, 0.f, 0.f, 0.f};
        if (q0 + r0 + it * 8 < Lq)
          z = *(const float4*)(Q + ((size_t)b * Lq + q0 + r0 + it * 8) * D + h * DH + c);
        rq[it] = z;
      }
    }
    for (int it = 0; it < 8; ++it) {
      _Float16* d = &sQ[(r0 + it * 8) * 72 + c];
      d[0] = (_Float16)(rq[it].x * scale); d[1] = (_Float16)(rq[it].y * scale);
      d[2] = (_Float16)(rq[it].z * scale); d[3] = (_Float16)(rq[it].w * scale);
    }
  }

  float rmax[8], rsum[8];
  v8f oacc[4] = {};
  for (int i = 0; i < 8; ++i) { rmax[i] = -1e30f; rsum[i] = 0.f; }

  const int qrow_base = q0 + wave * 16 + hi * 8;  // slot i -> query qrow_base+i
  int kmax = causal ? ((q0 + 64 < Lk) ? q0 + 64 : Lk) : Lk;
  int ktiles = (kmax + 31) >> 5;

  for (int kt = 0; kt < ktiles; ++kt) {
    // ---- stage K [key][dh] and V transposed [dh][key] (batched loads) ----
    {
      const int r0 = tid >> 4;        // key base, +8 per it
      const int c  = (tid & 15) * 4;  // dh
      float4 rk[4], rv[4];
      if (kt * 32 + 31 < Lk) {
        for (int it = 0; it < 4; ++it) {
          size_t base = ((size_t)b * Lk + kt * 32 + r0 + it * 8) * D + h * DH + c;
          rk[it] = *(const float4*)(K + base);
          rv[it] = *(const float4*)(V + base);
        }
      } else {
        for (int it = 0; it < 4; ++it) {
          float4 zk = {0.f, 0.f, 0.f, 0.f}, zv = {0.f, 0.f, 0.f, 0.f};
          if (kt * 32 + r0 + it * 8 < Lk) {
            size_t base = ((size_t)b * Lk + kt * 32 + r0 + it * 8) * D + h * DH + c;
            zk = *(const float4*)(K + base);
            zv = *(const float4*)(V + base);
          }
          rk[it] = zk; rv[it] = zv;
        }
      }
      for (int it = 0; it < 4; ++it) {
        int r = r0 + it * 8;
        _Float16* dk = &sK[r * 72 + c];
        dk[0] = (_Float16)rk[it].x; dk[1] = (_Float16)rk[it].y;
        dk[2] = (_Float16)rk[it].z; dk[3] = (_Float16)rk[it].w;
        sV[(c + 0) * 40 + r] = (_Float16)rv[it].x;
        sV[(c + 1) * 40 + r] = (_Float16)rv[it].y;
        sV[(c + 2) * 40 + r] = (_Float16)rv[it].z;
        sV[(c + 3) * 40 + r] = (_Float16)rv[it].w;
      }
    }
    __syncthreads();

    bool skip = (causal != 0) && (kt * 32 > q0 + wave * 16 + 15);  // wave-uniform
    if (!skip) {
      // ---- scores: S(16q x 32key) = Q(16x64) K^T(64x32), k-split over dh ----
      v8f s[2] = {};
      for (int step = 0; step < 2; ++step) {
        const v8h* a0 = (const v8h*)&sQ[(wave * 16 + lm) * 72 + step * 32 + hi * 8];
        const v8h* a1 = (const v8h*)&sQ[(wave * 16 + lm) * 72 + step * 32 + 16 + hi * 8];
        v16h af = SHUF16(*a0, *a1);
        for (int t = 0; t < 2; ++t) {
          const v8h* b0 = (const v8h*)&sK[(t * 16 + lm) * 72 + step * 32 + hi * 8];
          const v8h* b1 = (const v8h*)&sK[(t * 16 + lm) * 72 + step * 32 + 16 + hi * 8];
          v16h bf = SHUF16(*b0, *b1);
          s[t] = WMMA_F16(af, bf, s[t]);
        }
      }
      // ---- masking (key bound + causal) ----
      for (int t = 0; t < 2; ++t) {
        int key = kt * 32 + t * 16 + lm;
        for (int i = 0; i < 8; ++i) {
          if (key >= Lk || (causal && key > qrow_base + i)) s[t][i] = -1e30f;
        }
      }
      // ---- online softmax: row reductions across the 16-lane key groups ----
      float mnew[8], corr[8], tot[8];
      for (int i = 0; i < 8; ++i) mnew[i] = fmaxf(s[0][i], s[1][i]);
      for (int off = 1; off < 16; off <<= 1)
        for (int i = 0; i < 8; ++i)
          mnew[i] = fmaxf(mnew[i], __shfl_xor(mnew[i], off, 32));
      for (int i = 0; i < 8; ++i) {
        float m = fmaxf(rmax[i], mnew[i]);
        corr[i] = __expf(rmax[i] - m);
        rmax[i] = m;
      }
      for (int i = 0; i < 8; ++i) {
        float p0 = __expf(s[0][i] - rmax[i]);
        float p1 = __expf(s[1][i] - rmax[i]);
        s[0][i] = p0; s[1][i] = p1;
        tot[i] = p0 + p1;
      }
      for (int off = 1; off < 16; off <<= 1)
        for (int i = 0; i < 8; ++i)
          tot[i] += __shfl_xor(tot[i], off, 32);
      for (int i = 0; i < 8; ++i) rsum[i] = rsum[i] * corr[i] + tot[i];
      for (int nt = 0; nt < 4; ++nt)
        for (int i = 0; i < 8; ++i) oacc[nt][i] *= corr[i];

      // ---- relay P through per-wave LDS tile into A-fragment layout ----
      _Float16* pp = &sP[wave * 640];
      for (int t = 0; t < 2; ++t)
        for (int i = 0; i < 8; ++i)
          pp[(i + hi * 8) * 40 + t * 16 + lm] = (_Float16)s[t][i];
      const v8h* p0v = (const v8h*)&pp[lm * 40 + hi * 8];
      const v8h* p1v = (const v8h*)&pp[lm * 40 + 16 + hi * 8];
      v16h ap = SHUF16(*p0v, *p1v);

      // ---- O(16q x 64dh) += P(16x32) V(32x64) ----
      for (int nt = 0; nt < 4; ++nt) {
        const v8h* bv = (const v8h*)&sV[(nt * 16 + lm) * 40 + hi * 8];
        const v8h* bv1 = (const v8h*)&sV[(nt * 16 + lm) * 40 + 16 + hi * 8];
        v16h bf = SHUF16(*bv, *bv1);
        oacc[nt] = WMMA_F16(ap, bf, oacc[nt]);
      }
    }
    __syncthreads();
  }

  // write normalized output
  for (int i = 0; i < 8; ++i) {
    int q = qrow_base + i;
    if (q >= Lq) continue;
    float inv = 1.f / rsum[i];
    float* orow = O + ((size_t)b * Lq + q) * D + h * DH;
    for (int nt = 0; nt < 4; ++nt) orow[nt * 16 + lm] = oacc[nt][i] * inv;
  }
}

// ===========================================================================
// Embedding: circular conv1d (kernel 3) token embed + sinusoidal PE + time
// mark linear. grid = (D/256, L, B); one thread per (b, l, d).
// ===========================================================================
__global__ __launch_bounds__(256)
void embed_kernel(const float* __restrict__ x, const float* __restrict__ mark,
                  const float* __restrict__ tok_w, const float* __restrict__ time_w,
                  const float* __restrict__ time_b, float* __restrict__ out,
                  int L, int Cin, int Mk)
{
  const int Dm = 512;
  int d = threadIdx.x + blockIdx.x * 256;
  int l = blockIdx.y, b = blockIdx.z;
  float acc = time_b[d];
  for (int t = 0; t < 3; ++t) {
    int src = l - 1 + t;
    if (src < 0) src += L;
    if (src >= L) src -= L;
    const float* xr = x + ((size_t)b * L + src) * Cin;
    const float* wr = tok_w + (size_t)t * Cin * Dm + d;
    for (int c = 0; c < Cin; ++c) acc = fmaf(xr[c], wr[(size_t)c * Dm], acc);
  }
  const float* mr = mark + ((size_t)b * L + l) * Mk;
  for (int m = 0; m < Mk; ++m) acc = fmaf(mr[m], time_w[m * Dm + d], acc);
  // sinusoidal positional encoding
  float div = __expf(-(float)(d & ~1) * (__logf(10000.f) / (float)Dm));
  float ang = (float)l * div;
  acc += (d & 1) ? __cosf(ang) : __sinf(ang);
  out[((size_t)b * L + l) * Dm + d] = acc;
}

// ===========================================================================
// Fused residual + LayerNorm over D=512. One row per 256-thread block.
// out = LN(X + Yopt) * g + b   (Yopt may be null).
// ===========================================================================
__global__ __launch_bounds__(256)
void ln_kernel(const float* __restrict__ X, const float* __restrict__ Yopt,
               const float* __restrict__ g, const float* __restrict__ bb,
               float* __restrict__ out)
{
  const int D = 512;
  __shared__ float red[16];
  size_t row = blockIdx.x;
  int tid = threadIdx.x;
  const float* xr = X + row * D;
  float v0 = xr[tid], v1 = xr[tid + 256];
  if (Yopt) { const float* yr = Yopt + row * D; v0 += yr[tid]; v1 += yr[tid + 256]; }
  float s = v0 + v1;
  for (int off = 16; off > 0; off >>= 1) s += __shfl_xor(s, off, 32);
  if ((tid & 31) == 0) red[tid >> 5] = s;
  __syncthreads();
  float tot = 0.f;
  for (int i = 0; i < 8; ++i) tot += red[i];
  float mean = tot * (1.f / 512.f);
  float d0 = v0 - mean, d1 = v1 - mean;
  float s2 = d0 * d0 + d1 * d1;
  for (int off = 16; off > 0; off >>= 1) s2 += __shfl_xor(s2, off, 32);
  if ((tid & 31) == 0) red[8 + (tid >> 5)] = s2;
  __syncthreads();
  float tv = 0.f;
  for (int i = 0; i < 8; ++i) tv += red[8 + i];
  float rstd = rsqrtf(tv * (1.f / 512.f) + 1e-5f);
  out[row * D + tid]       = d0 * rstd * g[tid] + bb[tid];
  out[row * D + tid + 256] = d1 * rstd * g[tid + 256] + bb[tid + 256];
}

// ===========================================================================
// Final projection (N=21 — too thin for WMMA tiles) with last-PRED_LEN slice.
// out[b, t, c] = dec[b, Ld-PLEN+t, :] . proj_w[:, c] + proj_b[c]
// ===========================================================================
__global__ __launch_bounds__(256)
void proj_kernel(const float* __restrict__ X, const float* __restrict__ W,
                 const float* __restrict__ bias, float* __restrict__ out,
                 int Bn, int Ld, int PLEN, int Cout)
{
  const int D = 512;
  int idx = blockIdx.x * 256 + threadIdx.x;
  int total = Bn * PLEN * Cout;
  if (idx >= total) return;
  int c = idx % Cout;
  int t = (idx / Cout) % PLEN;
  int b = idx / (Cout * PLEN);
  const float* xr = X + ((size_t)b * Ld + (Ld - PLEN) + t) * D;
  float acc = bias[c];
  for (int d = 0; d < D; ++d) acc = fmaf(xr[d], W[(size_t)d * Cout + c], acc);
  out[idx] = acc;
}

// ===========================================================================
// Host orchestration
// ===========================================================================
static inline void run_gemm(const float* A, const float* B, const float* bias,
                            float* C, int M, int N, int K, int act, hipStream_t s)
{
  dim3 grid(N / 128, M / 128);   // all call sites: M%128==0, N%128==0
  gemm_f16_kernel<<<grid, dim3(256), 0, s>>>(A, B, bias, C, M, N, K, act);
}

extern "C" void kernel_launch(void* const* d_in, const int* in_sizes, int n_in,
                              void* d_out, int out_size, void* d_ws, size_t ws_size,
                              hipStream_t stream)
{
  (void)in_sizes; (void)n_in; (void)out_size; (void)ws_size;

  const float* x_enc      = (const float*)d_in[0];
  const float* x_mark_enc = (const float*)d_in[1];
  const float* x_dec      = (const float*)d_in[2];
  const float* x_mark_dec = (const float*)d_in[3];
  const float* enc_tok_w  = (const float*)d_in[4];
  const float* enc_time_w = (const float*)d_in[5];
  const float* enc_time_b = (const float*)d_in[6];
  const float* enc_attn_w = (const float*)d_in[7];
  const float* enc_attn_b = (const float*)d_in[8];
  const float* enc_ffn_w1 = (const float*)d_in[9];
  const float* enc_ffn_b1 = (const float*)d_in[10];
  const float* enc_ffn_w2 = (const float*)d_in[11];
  const float* enc_ffn_b2 = (const float*)d_in[12];
  const float* enc_ln_g   = (const float*)d_in[13];
  const float* enc_ln_b   = (const float*)d_in[14];
  const float* enc_norm_g = (const float*)d_in[15];
  const float* enc_norm_b = (const float*)d_in[16];
  const float* dec_tok_w  = (const float*)d_in[17];
  const float* dec_time_w = (const float*)d_in[18];
  const float* dec_time_b = (const float*)d_in[19];
  const float* dec_attn_w = (const float*)d_in[20];
  const float* dec_attn_b = (const float*)d_in[21];
  const float* dec_ffn_w1 = (const float*)d_in[22];
  const float* dec_ffn_b1 = (const float*)d_in[23];
  const float* dec_ffn_w2 = (const float*)d_in[24];
  const float* dec_ffn_b2 = (const float*)d_in[25];
  const float* dec_ln_g   = (const float*)d_in[26];
  const float* dec_ln_b   = (const float*)d_in[27];
  const float* dec_norm_g = (const float*)d_in[28];
  const float* dec_norm_b = (const float*)d_in[29];
  const float* proj_w     = (const float*)d_in[30];
  const float* proj_b     = (const float*)d_in[31];

  const int Bn = 8, Lenc = 720, Ldec = 672, D = 512, FF = 2048, H = 8;
  const int PLEN = 336, CIN = 21, MK = 4, COUT = 21;
  const int MR = Bn * Lenc;   // 5760 encoder rows (45 * 128)
  const int DR = Bn * Ldec;   // 5376 decoder rows (42 * 128)
  const size_t DD = (size_t)D * D;
  const size_t ENC_SZ = (size_t)MR * D;

  float* ws   = (float*)d_ws;
  float* bufX = ws;                 // activations
  float* bufY = bufX + ENC_SZ;      // attn out / LN temp
  float* bufQ = bufY + ENC_SZ;
  float* bufK = bufQ + ENC_SZ;
  float* bufV = bufK + ENC_SZ;
  float* bufE = bufV + ENC_SZ;      // persistent encoder output
  float* bufH = bufE + ENC_SZ;      // FFN hidden
  // total: 6*ENC_SZ + MR*FF floats ~= 118 MB

  // ---------------- Encoder ----------------
  embed_kernel<<<dim3(2, Lenc, Bn), 256, 0, stream>>>(
      x_enc, x_mark_enc, enc_tok_w, enc_time_w, enc_time_b, bufX, Lenc, CIN, MK);

  for (int l = 0; l < 2; ++l) {
    const float* aw = enc_attn_w + (size_t)l * 4 * DD;
    const float* ab = enc_attn_b + (size_t)l * 4 * D;
    run_gemm(bufX, aw + 0 * DD, ab + 0 * D, bufQ, MR, D, D, 0, stream);
    run_gemm(bufX, aw + 1 * DD, ab + 1 * D, bufK, MR, D, D, 0, stream);
    run_gemm(bufX, aw + 2 * DD, ab + 2 * D, bufV, MR, D, D, 0, stream);
    attn_kernel<<<dim3((Lenc + 63) / 64, H, Bn), 128, 0, stream>>>(
        bufQ, bufK, bufV, bufY, Lenc, Lenc, 0);
    run_gemm(bufY, aw + 3 * DD, ab + 3 * D, bufK, MR, D, D, 0, stream);
    ln_kernel<<<MR, 256, 0, stream>>>(bufX, bufK,
        enc_ln_g + (size_t)(l * 2 + 0) * D, enc_ln_b + (size_t)(l * 2 + 0) * D, bufY);
    run_gemm(bufY, enc_ffn_w1 + (size_t)l * D * FF, enc_ffn_b1 + (size_t)l * FF,
             bufH, MR, FF, D, 1, stream);  // fused GELU
    run_gemm(bufH, enc_ffn_w2 + (size_t)l * FF * D, enc_ffn_b2 + (size_t)l * D,
             bufQ, MR, D, FF, 0, stream);
    ln_kernel<<<MR, 256, 0, stream>>>(bufY, bufQ,
        enc_ln_g + (size_t)(l * 2 + 1) * D, enc_ln_b + (size_t)(l * 2 + 1) * D, bufX);
  }
  ln_kernel<<<MR, 256, 0, stream>>>(bufX, nullptr, enc_norm_g, enc_norm_b, bufE);

  // ---------------- Decoder (1 layer) ----------------
  embed_kernel<<<dim3(2, Ldec, Bn), 256, 0, stream>>>(
      x_dec, x_mark_dec, dec_tok_w, dec_time_w, dec_time_b, bufX, Ldec, CIN, MK);

  {
    const float* aw = dec_attn_w;  // (8, D, D)
    const float* ab = dec_attn_b;  // (8, D)
    // self-attention (causal)
    run_gemm(bufX, aw + 0 * DD, ab + 0 * D, bufQ, DR, D, D, 0, stream);
    run_gemm(bufX, aw + 1 * DD, ab + 1 * D, bufK, DR, D, D, 0, stream);
    run_gemm(bufX, aw + 2 * DD, ab + 2 * D, bufV, DR, D, D, 0, stream);
    attn_kernel<<<dim3((Ldec + 63) / 64, H, Bn), 128, 0, stream>>>(
        bufQ, bufK, bufV, bufY, Ldec, Ldec, 1);
    run_gemm(bufY, aw + 3 * DD, ab + 3 * D, bufK, DR, D, D, 0, stream);
    ln_kernel<<<DR, 256, 0, stream>>>(bufX, bufK,
        dec_ln_g + 0 * D, dec_ln_b + 0 * D, bufY);   // x -> bufY
    // cross-attention (queries from decoder, K/V from encoder output)
    run_gemm(bufY, aw + 4 * DD, ab + 4 * D, bufQ, DR, D, D, 0, stream);
    run_gemm(bufE, aw + 5 * DD, ab + 5 * D, bufK, MR, D, D, 0, stream);
    run_gemm(bufE, aw + 6 * DD, ab + 6 * D, bufV, MR, D, D, 0, stream);
    attn_kernel<<<dim3((Ldec + 63) / 64, H, Bn), 128, 0, stream>>>(
        bufQ, bufK, bufV, bufX, Ldec, Lenc, 0);
    run_gemm(bufX, aw + 7 * DD, ab + 7 * D, bufK, DR, D, D, 0, stream);
    ln_kernel<<<DR, 256, 0, stream>>>(bufY, bufK,
        dec_ln_g + 1 * D, dec_ln_b + 1 * D, bufX);   // x -> bufX
    // FFN
    run_gemm(bufX, dec_ffn_w1, dec_ffn_b1, bufH, DR, FF, D, 1, stream);
    run_gemm(bufH, dec_ffn_w2, dec_ffn_b2, bufQ, DR, D, FF, 0, stream);
    ln_kernel<<<DR, 256, 0, stream>>>(bufX, bufQ,
        dec_ln_g + 2 * D, dec_ln_b + 2 * D, bufY);
  }
  ln_kernel<<<DR, 256, 0, stream>>>(bufY, nullptr, dec_norm_g, dec_norm_b, bufX);

  // ---------------- Projection + slice ----------------
  {
    int total = Bn * PLEN * COUT;
    proj_kernel<<<(total + 255) / 256, 256, 0, stream>>>(
        bufX, proj_w, proj_b, (float*)d_out, Bn, Ldec, PLEN, COUT);
  }
}
